// TransposedAttention_fusion_50766513439206
// MI455X (gfx1250) — compile-verified, matrix operand order checked by reading, not run on hardware
//
#include <hip/hip_runtime.h>
#include <math.h>

typedef __attribute__((ext_vector_type(16))) _Float16 v16h;
typedef __attribute__((ext_vector_type(8)))  _Float16 h8;
typedef __attribute__((ext_vector_type(8)))  float    v8f;

#define BM 128
#define BN 128
#define BK 32
#define LDS_STRIDE 40   // halfs per LDS row: BK + 8 pad -> 80 bytes, 16B aligned
#define NTHREADS 256

enum { EPI_NONE = 0, EPI_BIAS = 1, EPI_BN = 2 };

// Y[m][n] = sum_k A[m][k] * B[k][n]   (BT=false: Bsrc indexed [k][n], ldb = row stride)
//                                      (BT=true : Bsrc indexed [n][k], ldb = row stride)
// Per-z offsets: off = (z % zmod)*stride1 + (z / zmod)*stride2   (zmod==1 -> off = z*stride2)
template<int EPI, bool BT>
__global__ __launch_bounds__(NTHREADS)
void gemm_wmma_f16(const float* __restrict__ A, const float* __restrict__ Bsrc,
                   float* __restrict__ Y,
                   int lda, int ldb, int ldc,
                   long long sA1, long long sA2,
                   long long sB1, long long sB2,
                   long long sY1, long long sY2,
                   int zmod, int K,
                   const float* __restrict__ bias,
                   const float* __restrict__ bnw,
                   const float* __restrict__ bnb)
{
    __shared__ _Float16 As[BM * LDS_STRIDE];
    __shared__ _Float16 Bs[BN * LDS_STRIDE];

    const int z  = blockIdx.z;
    const int z1 = (zmod > 1) ? (z % zmod) : 0;
    const int z2 = (zmod > 1) ? (z / zmod) : z;
    A    += z1 * sA1 + z2 * sA2;
    Bsrc += z1 * sB1 + z2 * sB2;
    Y    += z1 * sY1 + z2 * sY2;

    const int m0 = blockIdx.y * BM;
    const int n0 = blockIdx.x * BN;

    const int t     = threadIdx.x;
    const int lane  = t & 31;
    const int wave  = t >> 5;
    const int wm    = wave >> 1;          // 0..3  (rows of waves)
    const int wn    = wave & 1;           // 0..1  (cols of waves)
    const int lrow  = lane & 15;
    const int khalf = lane >> 4;          // 0 or 1
    const int kbase = khalf * 8;

    v8f acc[2][4];
#pragma unroll
    for (int i = 0; i < 2; ++i)
#pragma unroll
        for (int j = 0; j < 4; ++j)
#pragma unroll
            for (int e = 0; e < 8; ++e) acc[i][j][e] = 0.0f;

    for (int k0 = 0; k0 < K; k0 += BK) {
        // ---- stage A tile (BM x BK), f32 -> f16, K-contiguous in LDS ----
#pragma unroll
        for (int i = 0; i < 4; ++i) {
            int q  = t + i * NTHREADS;          // 0..1023 quads
            int m  = q >> 3;                    // 0..127
            int kq = (q & 7) << 2;              // 0,4,..,28
            float4 v = *(const float4*)(A + (size_t)(m0 + m) * lda + k0 + kq);
            _Float16* dst = As + m * LDS_STRIDE + kq;
            dst[0] = (_Float16)v.x; dst[1] = (_Float16)v.y;
            dst[2] = (_Float16)v.z; dst[3] = (_Float16)v.w;
        }
        // ---- stage B tile -> Bs[n][k] (K-contiguous) ----
        if (BT) {
#pragma unroll
            for (int i = 0; i < 4; ++i) {
                int q  = t + i * NTHREADS;
                int n  = q >> 3;
                int kq = (q & 7) << 2;
                float4 v = *(const float4*)(Bsrc + (size_t)(n0 + n) * ldb + k0 + kq);
                _Float16* dst = Bs + n * LDS_STRIDE + kq;
                dst[0] = (_Float16)v.x; dst[1] = (_Float16)v.y;
                dst[2] = (_Float16)v.z; dst[3] = (_Float16)v.w;
            }
        } else {
#pragma unroll
            for (int i = 0; i < 4; ++i) {
                int q  = t + i * NTHREADS;
                int k  = q >> 5;                 // 0..31
                int nq = (q & 31) << 2;          // 0..124
                float4 v = *(const float4*)(Bsrc + (size_t)(k0 + k) * ldb + n0 + nq);
                Bs[(nq + 0) * LDS_STRIDE + k] = (_Float16)v.x;
                Bs[(nq + 1) * LDS_STRIDE + k] = (_Float16)v.y;
                Bs[(nq + 2) * LDS_STRIDE + k] = (_Float16)v.z;
                Bs[(nq + 3) * LDS_STRIDE + k] = (_Float16)v.w;
            }
        }
        __syncthreads();

        // ---- load fragments (two ds_load_b128 each, matches 16-bit A layout) ----
        v16h afrag[2], bfrag[4];
#pragma unroll
        for (int tm = 0; tm < 2; ++tm) {
            const _Float16* p = As + (wm * 32 + tm * 16 + lrow) * LDS_STRIDE + kbase;
            h8 lo = *(const h8*)p;               // K = kbase .. kbase+7
            h8 hi = *(const h8*)(p + 16);        // K = kbase+16 .. kbase+23
#pragma unroll
            for (int e = 0; e < 8; ++e) { afrag[tm][e] = lo[e]; afrag[tm][8 + e] = hi[e]; }
        }
#pragma unroll
        for (int tn = 0; tn < 4; ++tn) {
            const _Float16* p = Bs + (wn * 64 + tn * 16 + lrow) * LDS_STRIDE + kbase;
            h8 lo = *(const h8*)p;
            h8 hi = *(const h8*)(p + 16);
#pragma unroll
            for (int e = 0; e < 8; ++e) { bfrag[tn][e] = lo[e]; bfrag[tn][8 + e] = hi[e]; }
        }

        // ---- 8 WMMAs per K-step ----
#pragma unroll
        for (int tm = 0; tm < 2; ++tm)
#pragma unroll
            for (int tn = 0; tn < 4; ++tn)
                acc[tm][tn] = __builtin_amdgcn_wmma_f32_16x16x32_f16(
                    false, afrag[tm], false, bfrag[tn],
                    (short)0, acc[tm][tn], false, false);
        __syncthreads();
    }

    // ---- epilogue: D layout: M = v + 8*khalf, N = lrow ----
    const float bnrs = 0.9999950000374997f;   // rsqrt(1 + 1e-5)
#pragma unroll
    for (int tm = 0; tm < 2; ++tm) {
#pragma unroll
        for (int tn = 0; tn < 4; ++tn) {
            int col = n0 + wn * 64 + tn * 16 + lrow;
#pragma unroll
            for (int v = 0; v < 8; ++v) {
                int row = m0 + wm * 32 + tm * 16 + khalf * 8 + v;
                float val = acc[tm][tn][v];
                if (EPI == EPI_BIAS) {
                    val += bias[row];
                } else if (EPI == EPI_BN) {
                    float s = bnw[row] * bnrs;
                    val = (val + bias[row]) * s + bnb[row];
                }
                Y[(size_t)row * ldc + col] = val;
            }
        }
    }
}

// L2-normalize 1024-wide rows in place. Rows 0..4095 -> q, rows 4096..8191 -> k half of kv.
__global__ __launch_bounds__(256)
void l2norm_rows(float* __restrict__ q, float* __restrict__ kv)
{
    int r = blockIdx.x;
    float* row;
    if (r < 4096) {
        row = q + (size_t)r * 1024;
    } else {
        int i = r - 4096;
        int b = i >> 11, c = i & 2047;
        row = kv + (size_t)b * 4194304 + (size_t)c * 1024;   // k = rows [0,2048) of kv slab
    }
    int t = threadIdx.x;
    float s = 0.0f;
    for (int j = t; j < 1024; j += 256) { float x = row[j]; s += x * x; }
    __shared__ float red[256];
    red[t] = s; __syncthreads();
    for (int off = 128; off > 0; off >>= 1) {
        if (t < off) red[t] += red[t + off];
        __syncthreads();
    }
    float inv = 1.0f / fmaxf(sqrtf(red[0]), 1e-12f);
    for (int j = t; j < 1024; j += 256) row[j] *= inv;
}

// Temperature-scaled softmax over 256-wide rows, in place. 4096 rows = [b][head][256].
__global__ __launch_bounds__(256)
void softmax_rows(float* __restrict__ attn, const float* __restrict__ temp)
{
    int r = blockIdx.x;
    int head = (r >> 8) & 7;
    float* row = attn + (size_t)r * 256;
    int t = threadIdx.x;
    float v = row[t] * temp[head];
    __shared__ float red[256];
    red[t] = v; __syncthreads();
    for (int off = 128; off > 0; off >>= 1) {
        if (t < off) red[t] = fmaxf(red[t], red[t + off]);
        __syncthreads();
    }
    float m = red[0]; __syncthreads();
    float e = __expf(v - m);
    red[t] = e; __syncthreads();
    for (int off = 128; off > 0; off >>= 1) {
        if (t < off) red[t] += red[t + off];
        __syncthreads();
    }
    row[t] = e / red[0];
}

extern "C" void kernel_launch(void* const* d_in, const int* in_sizes, int n_in,
                              void* d_out, int out_size, void* d_ws, size_t ws_size,
                              hipStream_t stream)
{
    (void)in_sizes; (void)n_in; (void)out_size; (void)ws_size;

    const float* x1     = (const float*)d_in[0];
    const float* x2     = (const float*)d_in[1];
    const float* kv_w   = (const float*)d_in[2];
    const float* kv_b   = (const float*)d_in[3];
    const float* q_w    = (const float*)d_in[4];
    const float* q_b    = (const float*)d_in[5];
    const float* proj_w = (const float*)d_in[6];
    const float* proj_b = (const float*)d_in[7];
    const float* temp   = (const float*)d_in[8];
    const float* bn1_w  = (const float*)d_in[9];
    const float* bn1_b  = (const float*)d_in[10];
    const float* bn2_w  = (const float*)d_in[11];
    const float* bn2_b  = (const float*)d_in[12];
    float* out = (float*)d_out;

    float* ws   = (float*)d_ws;
    float* kv   = ws;                 // [2][4096][1024]  (k rows 0..2047, v rows 2048..4095)
    float* qn   = ws + 8388608;       // [2][2048][1024]
    float* attn = ws + 12582912;      // [2][8][256][256]
    float* ao   = ws + 13631488;      // [2][2048][1024]

    dim3 blk(NTHREADS);

    // 1) kv = BN1(x1 @ kv_w^T + kv_b)   M=4096 N=1024 K=2048, batched over b=2
    gemm_wmma_f16<EPI_BN, false><<<dim3(8, 32, 2), blk, 0, stream>>>(
        kv_w, x1, kv, 2048, 1024, 1024,
        0, 0,   0, 2048LL * 1024,   0, 4096LL * 1024,
        1, 2048, kv_b, bn1_w, bn1_b);

    // 2) q = BN2(x2 @ q_w^T + q_b)      M=2048 N=1024 K=2048
    gemm_wmma_f16<EPI_BN, false><<<dim3(8, 16, 2), blk, 0, stream>>>(
        q_w, x2, qn, 2048, 1024, 1024,
        0, 0,   0, 2048LL * 1024,   0, 2048LL * 1024,
        1, 2048, q_b, bn2_w, bn2_b);

    // 3) L2-normalize q rows and k rows (over n=1024), in place
    l2norm_rows<<<8192, blk, 0, stream>>>(qn, kv);

    // 4) attn = q @ k^T per (b,head)    M=256 N=256 K=1024, z = b*8+head
    gemm_wmma_f16<EPI_NONE, true><<<dim3(2, 2, 16), blk, 0, stream>>>(
        qn, kv, attn, 1024, 1024, 256,
        262144LL, 2097152LL,   262144LL, 4194304LL,   65536LL, 524288LL,
        8, 1024, nullptr, nullptr, nullptr);

    // 5) softmax(attn * temperature) over last dim
    softmax_rows<<<4096, blk, 0, stream>>>(attn, temp);

    // 6) ao = attn @ v per (b,head)     M=256 N=1024 K=256
    gemm_wmma_f16<EPI_NONE, false><<<dim3(8, 2, 16), blk, 0, stream>>>(
        attn, kv + 2097152, ao, 256, 1024, 1024,
        65536LL, 524288LL,   262144LL, 4194304LL,   262144LL, 2097152LL,
        8, 256, nullptr, nullptr, nullptr);

    // 7) out = ao @ proj_w^T + proj_b   M=2048 N=1024 K=2048, batched over b=2
    gemm_wmma_f16<EPI_BIAS, false><<<dim3(8, 16, 2), blk, 0, stream>>>(
        proj_w, ao, out, 2048, 1024, 1024,
        0, 0,   0, 2048LL * 1024,   0, 2048LL * 1024,
        1, 2048, proj_b, nullptr, nullptr);
}
